// Llama4VisionAttention_86526411145417
// MI455X (gfx1250) — matrix operand order repaired and checked by reading, not compiled, not run on hardware
//
#include <hip/hip_runtime.h>
#include <hip/hip_bf16.h>

typedef __bf16 bf16_t;
typedef __attribute__((ext_vector_type(16))) __bf16 v16bf;
typedef __attribute__((ext_vector_type(8)))  float  v8f;
typedef int v4i_vec __attribute__((vector_size(16)));

#define NB 16
#define SS 576
#define EE 1408
#define HH 16
#define DD 88
#define DP 96
#define NTOK (NB*SS)     // 9216
#define E3 (3*EE)        // 4224

union FragBF { v16bf v; unsigned long long u[4]; };

__device__ __forceinline__ void wait_lds0() {
#if __has_builtin(__builtin_amdgcn_s_wait_dscnt)
    __builtin_amdgcn_s_wait_dscnt(0);
#else
    asm volatile("s_wait_dscnt 0" ::: "memory");
#endif
}

// ---- CDNA5 async global->LDS path (GLOBAL_LOAD_ASYNC_TO_LDS_B128, ASYNCcnt) ----
#if __has_builtin(__builtin_amdgcn_global_load_async_to_lds_b128)
#define HAVE_ASYNC_LDS 1
__device__ __forceinline__ void cp_async16(bf16_t* lds, const bf16_t* g) {
    __builtin_amdgcn_global_load_async_to_lds_b128(
        (__attribute__((address_space(1))) v4i_vec*)g,
        (__attribute__((address_space(3))) v4i_vec*)lds, 0, 0);
}
#endif

template <int N>
__device__ __forceinline__ void wait_async() {
#if __has_builtin(__builtin_amdgcn_s_wait_asynccnt)
    __builtin_amdgcn_s_wait_asynccnt(N);
#else
    if (N == 0)      asm volatile("s_wait_asynccnt 0x0" ::: "memory");
    else if (N == 4) asm volatile("s_wait_asynccnt 0x4" ::: "memory");
    else             asm volatile("s_wait_asynccnt 0x0" ::: "memory");
#endif
}

// ---------------------------------------------------------------------------
// 1) fp32 -> bf16 convert
// ---------------------------------------------------------------------------
__global__ void cvt_bf16_kernel(const float* __restrict__ in, bf16_t* __restrict__ out, int n) {
    int i = blockIdx.x * blockDim.x + threadIdx.x;
    if (i < n) out[i] = (bf16_t)in[i];
}

// 2) transpose + convert: out[n*K + k] = bf16(in[k*N + n])   (in is KxN row-major)
__global__ void transpose_bf16_kernel(const float* __restrict__ in, bf16_t* __restrict__ out,
                                      int K, int N) {
    long long idx = (long long)blockIdx.x * blockDim.x + threadIdx.x;
    long long total = (long long)K * N;
    if (idx >= total) return;
    int k = (int)(idx % K);
    int n = (int)(idx / K);
    out[idx] = (bf16_t)in[(long long)k * N + n];
}

// ---------------------------------------------------------------------------
// 3) GEMM: C[M,N] = A[M,K](bf16) * BT[N,K](bf16)^T + bias
//    128x128 block tile, 256 threads = 8 waves (2 M-waves x 4 N-waves),
//    each wave does a 64x32 sub-tile = 4x2 v_wmma_f32_16x16x32_bf16.
//    LDS tiles double-buffered via async global->LDS copies when available.
// ---------------------------------------------------------------------------
#define GSTR 40   // LDS row stride in elements (32 + 8 pad, keeps 8B alignment)

__global__ __launch_bounds__(256) void gemm_bf16_wmma(
    const bf16_t* __restrict__ A, const bf16_t* __restrict__ BT,
    const float* __restrict__ bias,
    bf16_t* __restrict__ outb, float* __restrict__ outf,
    int M, int N, int K)
{
    __shared__ bf16_t As[2][128 * GSTR];
    __shared__ bf16_t Bs[2][128 * GSTR];

    const int tid  = threadIdx.x;
    const int lane = tid & 31;
    const int wave = tid >> 5;
    const int m0 = blockIdx.y * 128;
    const int n0 = blockIdx.x * 128;
    const int wm = (wave >> 2) * 64;   // 0 / 64
    const int wn = (wave & 3) * 32;    // 0 / 32 / 64 / 96
    const int r    = lane & 15;
    const int hi   = lane >> 4;        // 0: lanes 0-15, 1: lanes 16-31
    const int koffA = hi ? 8 : 0;
    const int koffB = hi ? 8 : 0;

    const int lrow = tid >> 1;         // 0..127
    const int lcol = (tid & 1) * 16;   // 0 / 16 (elements)

    const bf16_t* gA = A  + (size_t)(m0 + lrow) * K + lcol;
    const bf16_t* gB = BT + (size_t)(n0 + lrow) * K + lcol;
    const int ldst = lrow * GSTR + lcol;

    v8f acc[4][2];
    v8f zero = {0.f,0.f,0.f,0.f,0.f,0.f,0.f,0.f};
#pragma unroll
    for (int i = 0; i < 4; ++i)
#pragma unroll
        for (int j = 0; j < 2; ++j) acc[i][j] = zero;

    const int nk = K / 32;

#if defined(HAVE_ASYNC_LDS)
    // prologue: async-stage tile 0 into buffer 0
    cp_async16(&As[0][ldst],     gA);
    cp_async16(&As[0][ldst + 8], gA + 8);
    cp_async16(&Bs[0][ldst],     gB);
    cp_async16(&Bs[0][ldst + 8], gB + 8);
#endif

    for (int it = 0; it < nk; ++it) {
        const int cur = it & 1;
        const int k0  = it * 32;
#if defined(HAVE_ASYNC_LDS)
        if (it + 1 < nk) {   // stage next tile into other buffer, then wait for current
            const int nxt = cur ^ 1;
            const int kn  = k0 + 32;
            cp_async16(&As[nxt][ldst],     gA + kn);
            cp_async16(&As[nxt][ldst + 8], gA + kn + 8);
            cp_async16(&Bs[nxt][ldst],     gB + kn);
            cp_async16(&Bs[nxt][ldst + 8], gB + kn + 8);
            wait_async<4>();   // async in-order: current tile's 4 copies complete
        } else {
            wait_async<0>();
        }
        __syncthreads();       // all waves' copies for buffer `cur` complete
#else
        {
            const uint4* ga = (const uint4*)(gA + k0);
            const uint4* gb = (const uint4*)(gB + k0);
            uint4 a0 = ga[0], a1 = ga[1];
            uint4 b0 = gb[0], b1 = gb[1];
            if (k0 + 32 < K) {  // speculative prefetch -> global_prefetch_b8
                __builtin_prefetch(gA + k0 + 32, 0, 1);
                __builtin_prefetch(gB + k0 + 32, 0, 1);
            }
            __syncthreads();
            *(uint4*)(&As[cur][ldst])     = a0;
            *(uint4*)(&As[cur][ldst + 8]) = a1;
            *(uint4*)(&Bs[cur][ldst])     = b0;
            *(uint4*)(&Bs[cur][ldst + 8]) = b1;
            __syncthreads();
        }
#endif
        FragBF aF[4], bF[2];
#pragma unroll
        for (int i = 0; i < 4; ++i) {
            const bf16_t* p = &As[cur][(wm + i * 16 + r) * GSTR + koffA];
            aF[i].u[0] = *(const unsigned long long*)(p);
            aF[i].u[1] = *(const unsigned long long*)(p + 4);
            aF[i].u[2] = *(const unsigned long long*)(p + 16);
            aF[i].u[3] = *(const unsigned long long*)(p + 20);
        }
#pragma unroll
        for (int j = 0; j < 2; ++j) {
            const bf16_t* p = &Bs[cur][(wn + j * 16 + r) * GSTR + koffB];
            bF[j].u[0] = *(const unsigned long long*)(p);
            bF[j].u[1] = *(const unsigned long long*)(p + 4);
            bF[j].u[2] = *(const unsigned long long*)(p + 16);
            bF[j].u[3] = *(const unsigned long long*)(p + 20);
        }
#pragma unroll
        for (int i = 0; i < 4; ++i)
#pragma unroll
            for (int j = 0; j < 2; ++j)
                acc[i][j] = __builtin_amdgcn_wmma_f32_16x16x32_bf16(
                    false, aF[i].v, false, bF[j].v, (short)0, acc[i][j], false, false);
        __syncthreads();   // everyone done reading buffer `cur` before it is re-staged
    }

    // epilogue: C tile element e -> row = base + e + 8*hi, col = n-base + r
#pragma unroll
    for (int i = 0; i < 4; ++i) {
#pragma unroll
        for (int j = 0; j < 2; ++j) {
            int rbase = m0 + wm + i * 16 + hi * 8;
            int col   = n0 + wn + j * 16 + r;
            float bv  = bias ? bias[col] : 0.f;
            if (outf) {
#pragma unroll
                for (int e = 0; e < 8; ++e)
                    outf[(size_t)(rbase + e) * N + col] = acc[i][j][e] + bv;
            } else {
#pragma unroll
                for (int e = 0; e < 8; ++e)
                    outb[(size_t)(rbase + e) * N + col] = (bf16_t)(acc[i][j][e] + bv);
            }
        }
    }
}

// ---------------------------------------------------------------------------
// 4) RoPE + repack:  qkv(bf16, [NTOK, 3E]) ->
//    Qb (BH, S, 96) bf16  (rope + 1/sqrt(88) baked in, dims 88..95 zero)
//    Kb (BH, S, 96) bf16  (rope, padded)
//    VT (BH, 96, S) bf16  (transposed per head, padded)
// ---------------------------------------------------------------------------
__global__ void rope_pack_kernel(const bf16_t* __restrict__ qkv,
                                 bf16_t* __restrict__ Qb,
                                 bf16_t* __restrict__ Kb,
                                 bf16_t* __restrict__ VT)
{
    long long idx = (long long)blockIdx.x * blockDim.x + threadIdx.x;
    const long long total = (long long)NB * SS * HH * 48;
    if (idx >= total) return;
    int i = (int)(idx % 48);
    int h = (int)((idx / 48) % HH);
    int s = (int)((idx / (48 * HH)) % SS);
    int b = (int)(idx / (48LL * HH * SS));
    int bh = b * HH + h;
    size_t qrow = ((size_t)bh * SS + s) * DP;
    int d0 = 2 * i;

    if (i >= 44) {  // pad dims 88..95
        bf16_t z = (bf16_t)0.f;
        Qb[qrow + d0] = z;  Qb[qrow + d0 + 1] = z;
        Kb[qrow + d0] = z;  Kb[qrow + d0 + 1] = z;
        VT[((size_t)bh * DP + d0) * SS + s]     = z;
        VT[((size_t)bh * DP + d0 + 1) * SS + s] = z;
        return;
    }

    size_t tok = (size_t)b * SS + s;
    const bf16_t* qp = qkv + tok * E3 + h * DD + d0;
    float q0 = (float)qp[0],        q1 = (float)qp[1];
    float k0 = (float)qp[EE],       k1 = (float)qp[EE + 1];
    float v0 = (float)qp[2 * EE],   v1 = (float)qp[2 * EE + 1];

    // vision rope: grid=24x24; fx=s%24+1, fy=s/24+1; rf[j]=1e4^(-j/22)
    int  j = (i < 22) ? i : (i - 22);
    float f = (i < 22) ? (float)(s % 24 + 1) : (float)(s / 24 + 1);
    float rf = __expf(-((float)j / 22.0f) * 9.210340371976184f); // ln(10000)
    float ang = f * rf;
    float c = __cosf(ang), sn = __sinf(ang);
    const float scaling = 0.10660035817780521f;  // 88^-0.5

    Qb[qrow + d0]     = (bf16_t)((q0 * c - q1 * sn) * scaling);
    Qb[qrow + d0 + 1] = (bf16_t)((q0 * sn + q1 * c) * scaling);
    Kb[qrow + d0]     = (bf16_t)(k0 * c - k1 * sn);
    Kb[qrow + d0 + 1] = (bf16_t)(k0 * sn + k1 * c);
    VT[((size_t)bh * DP + d0) * SS + s]     = (bf16_t)v0;
    VT[((size_t)bh * DP + d0 + 1) * SS + s] = (bf16_t)v1;
}

// ---------------------------------------------------------------------------
// 5) Flash attention: grid (BH=256, 9), block 128 (4 waves).
//    Wave w owns Q rows [blockIdx.y*64 + w*16, +16). Streams keys in chunks
//    of 32: scores = 2 N-tiles x 3 K-steps WMMA; online softmax with
//    shfl_xor row reductions; probs re-laid-out via per-wave LDS; PV = 6 WMMA.
//    K/V tiles staged with async global->LDS copies when available.
// ---------------------------------------------------------------------------
#define KTSTR 104  // 96+8
#define VTSTR 40   // 32+8

__global__ __launch_bounds__(128) void attn_fwd_kernel(
    const bf16_t* __restrict__ Qb, const bf16_t* __restrict__ Kb,
    const bf16_t* __restrict__ VTg, bf16_t* __restrict__ Ob)
{
    __shared__ bf16_t Kt[32 * KTSTR];        // 32 keys x 96 dims
    __shared__ bf16_t Vt[DP * VTSTR];        // 96 dims x 32 keys
    __shared__ bf16_t Pt[4][16 * VTSTR];     // per-wave 16x32 probs

    const int tid  = threadIdx.x;
    const int lane = tid & 31;
    const int wave = tid >> 5;
    const int bh   = blockIdx.x;
    const int b    = bh >> 4;
    const int h    = bh & 15;
    const int qbase = blockIdx.y * 64 + wave * 16;
    const int r    = lane & 15;
    const int hi   = lane >> 4;
    const int koffA = hi ? 8 : 0;
    const int koffB = hi ? 8 : 0;

    // Q fragments for this wave's 16-row strip (row = qbase + r)
    const bf16_t* qrow = Qb + ((size_t)bh * SS + qbase + r) * DP;
    FragBF aQ[3];
#pragma unroll
    for (int ks = 0; ks < 3; ++ks) {
        const bf16_t* p = qrow + ks * 32 + koffA;
        aQ[ks].u[0] = *(const unsigned long long*)(p);
        aQ[ks].u[1] = *(const unsigned long long*)(p + 4);
        aQ[ks].u[2] = *(const unsigned long long*)(p + 16);
        aQ[ks].u[3] = *(const unsigned long long*)(p + 20);
    }

    float mrow[8], lsum[8];
#pragma unroll
    for (int e = 0; e < 8; ++e) { mrow[e] = -3.0e38f; lsum[e] = 0.f; }
    v8f zero = {0.f,0.f,0.f,0.f,0.f,0.f,0.f,0.f};
    v8f Oacc[6];
#pragma unroll
    for (int dt = 0; dt < 6; ++dt) Oacc[dt] = zero;

    const int krow = tid >> 2, kseg = (tid & 3) * 24;

    for (int kc = 0; kc < SS / 32; ++kc) {
        const int s0 = kc * 32;
        __syncthreads();  // previous chunk's LDS reads done
#if defined(HAVE_ASYNC_LDS)
        {
            const bf16_t* g = Kb + ((size_t)bh * SS + s0 + krow) * DP + kseg;
            bf16_t* d = &Kt[krow * KTSTR + kseg];
            cp_async16(d, g); cp_async16(d + 8, g + 8); cp_async16(d + 16, g + 16);
            if (tid < DP) {
                const bf16_t* gv = VTg + ((size_t)bh * DP + tid) * SS + s0;
                bf16_t* dv = &Vt[tid * VTSTR];
                cp_async16(dv, gv);           cp_async16(dv + 8,  gv + 8);
                cp_async16(dv + 16, gv + 16); cp_async16(dv + 24, gv + 24);
            }
            wait_async<0>();
        }
#else
        {   // stage K chunk: 32 rows x 96 dims, 24 elems/thread
            const uint4* g = (const uint4*)(Kb + ((size_t)bh * SS + s0 + krow) * DP + kseg);
            uint4 x0 = g[0], x1 = g[1], x2 = g[2];
            uint4* d = (uint4*)(&Kt[krow * KTSTR + kseg]);
            d[0] = x0; d[1] = x1; d[2] = x2;
        }
        if (tid < DP) {  // stage V^T chunk: 96 dims x 32 keys
            const uint4* g = (const uint4*)(VTg + ((size_t)bh * DP + tid) * SS + s0);
            uint4 x0 = g[0], x1 = g[1], x2 = g[2], x3 = g[3];
            uint4* d = (uint4*)(&Vt[tid * VTSTR]);
            d[0] = x0; d[1] = x1; d[2] = x2; d[3] = x3;
        }
#endif
        __syncthreads();

        // scores: 2 key-tiles x 3 k-steps
        v8f sc[2] = {zero, zero};
#pragma unroll
        for (int nt = 0; nt < 2; ++nt) {
#pragma unroll
            for (int ks = 0; ks < 3; ++ks) {
                FragBF bK;
                const bf16_t* p = &Kt[(nt * 16 + r) * KTSTR + ks * 32 + koffB];
                bK.u[0] = *(const unsigned long long*)(p);
                bK.u[1] = *(const unsigned long long*)(p + 4);
                bK.u[2] = *(const unsigned long long*)(p + 16);
                bK.u[3] = *(const unsigned long long*)(p + 20);
                sc[nt] = __builtin_amdgcn_wmma_f32_16x16x32_bf16(
                    false, aQ[ks].v, false, bK.v, (short)0, sc[nt], false, false);
            }
        }

        // online softmax over this 32-key chunk
        float cm[8];
#pragma unroll
        for (int e = 0; e < 8; ++e) cm[e] = fmaxf(sc[0][e], sc[1][e]);
#pragma unroll
        for (int mask = 1; mask <= 8; mask <<= 1)
#pragma unroll
            for (int e = 0; e < 8; ++e)
                cm[e] = fmaxf(cm[e], __shfl_xor(cm[e], mask, 32));

        float corr[8], p0[8], p1[8], csum[8];
#pragma unroll
        for (int e = 0; e < 8; ++e) {
            float mn = fmaxf(mrow[e], cm[e]);
            corr[e] = __expf(mrow[e] - mn);
            p0[e] = __expf(sc[0][e] - mn);
            p1[e] = __expf(sc[1][e] - mn);
            csum[e] = p0[e] + p1[e];
            mrow[e] = mn;
        }
#pragma unroll
        for (int mask = 1; mask <= 8; mask <<= 1)
#pragma unroll
            for (int e = 0; e < 8; ++e)
                csum[e] += __shfl_xor(csum[e], mask, 32);
#pragma unroll
        for (int e = 0; e < 8; ++e) lsum[e] = lsum[e] * corr[e] + csum[e];
#pragma unroll
        for (int dt = 0; dt < 6; ++dt)
#pragma unroll
            for (int e = 0; e < 8; ++e) Oacc[dt][e] *= corr[e];

        // C-tile -> A-fragment relayout of probs via per-wave LDS
        bf16_t* pw = &Pt[wave][0];
#pragma unroll
        for (int e = 0; e < 8; ++e) {
            int prow = e + hi * 8;
            pw[prow * VTSTR + r]      = (bf16_t)p0[e];
            pw[prow * VTSTR + 16 + r] = (bf16_t)p1[e];
        }
        wait_lds0();  // same-wave LDS RAW (in-order per wave; explicit wait)

        FragBF aP;
        {
            const bf16_t* p = &Pt[wave][r * VTSTR + koffA];
            aP.u[0] = *(const unsigned long long*)(p);
            aP.u[1] = *(const unsigned long long*)(p + 4);
            aP.u[2] = *(const unsigned long long*)(p + 16);
            aP.u[3] = *(const unsigned long long*)(p + 20);
        }
#pragma unroll
        for (int dt = 0; dt < 6; ++dt) {
            FragBF bV;
            const bf16_t* p = &Vt[(dt * 16 + r) * VTSTR + koffB];
            bV.u[0] = *(const unsigned long long*)(p);
            bV.u[1] = *(const unsigned long long*)(p + 4);
            bV.u[2] = *(const unsigned long long*)(p + 16);
            bV.u[3] = *(const unsigned long long*)(p + 20);
            Oacc[dt] = __builtin_amdgcn_wmma_f32_16x16x32_bf16(
                false, aP.v, false, bV.v, (short)0, Oacc[dt], false, false);
        }
    }

    // normalize + store to Ob (B,S,H,88) = (NTOK, 1408)
    float inv[8];
#pragma unroll
    for (int e = 0; e < 8; ++e) inv[e] = 1.0f / lsum[e];
#pragma unroll
    for (int dt = 0; dt < 6; ++dt) {
        int d = dt * 16 + r;
        if (d < DD) {
#pragma unroll
            for (int e = 0; e < 8; ++e) {
                int s = qbase + e + hi * 8;
                Ob[((size_t)(b * SS + s)) * EE + h * DD + d] = (bf16_t)(Oacc[dt][e] * inv[e]);
            }
        }
    }
}

// ---------------------------------------------------------------------------
extern "C" void kernel_launch(void* const* d_in, const int* in_sizes, int n_in,
                              void* d_out, int out_size, void* d_ws, size_t ws_size,
                              hipStream_t stream) {
    const float* hs    = (const float*)d_in[0];   // (16,576,1408)
    const float* w_qkv = (const float*)d_in[1];   // (1408,4224)
    const float* b_qkv = (const float*)d_in[2];   // (4224,)
    const float* w_o   = (const float*)d_in[3];   // (1408,1408)
    const float* b_o   = (const float*)d_in[4];   // (1408,)
    float* out = (float*)d_out;

    char* ws = (char*)d_ws;
    size_t off = 0;
    auto alloc = [&](size_t bytes) { size_t o = off; off += (bytes + 255) & ~(size_t)255; return o; };
    bf16_t* Xb     = (bf16_t*)(ws + alloc((size_t)NTOK * EE * 2));
    bf16_t* WqkvT  = (bf16_t*)(ws + alloc((size_t)E3 * EE * 2));
    bf16_t* WoT    = (bf16_t*)(ws + alloc((size_t)EE * EE * 2));
    bf16_t* QKVb   = (bf16_t*)(ws + alloc((size_t)NTOK * E3 * 2));
    bf16_t* Qb     = (bf16_t*)(ws + alloc((size_t)NB * HH * SS * DP * 2));
    bf16_t* Kb     = (bf16_t*)(ws + alloc((size_t)NB * HH * SS * DP * 2));
    bf16_t* VT     = (bf16_t*)(ws + alloc((size_t)NB * HH * SS * DP * 2));
    bf16_t* Ob     = (bf16_t*)(ws + alloc((size_t)NTOK * EE * 2));
    (void)ws_size; (void)in_sizes; (void)n_in; (void)out_size;

    // 1) convert activations
    {
        int n = NTOK * EE;
        cvt_bf16_kernel<<<(n + 255) / 256, 256, 0, stream>>>(hs, Xb, n);
    }
    // 2) transposed bf16 weights
    {
        long long t1 = (long long)EE * E3;
        transpose_bf16_kernel<<<(unsigned)((t1 + 255) / 256), 256, 0, stream>>>(w_qkv, WqkvT, EE, E3);
        long long t2 = (long long)EE * EE;
        transpose_bf16_kernel<<<(unsigned)((t2 + 255) / 256), 256, 0, stream>>>(w_o, WoT, EE, EE);
    }
    // 3) QKV projection: (9216,1408) x (1408,4224) + b_qkv -> bf16
    {
        dim3 grid(E3 / 128, NTOK / 128);
        gemm_bf16_wmma<<<grid, 256, 0, stream>>>(Xb, WqkvT, b_qkv, QKVb, nullptr,
                                                 NTOK, E3, EE);
    }
    // 4) RoPE + repack
    {
        long long total = (long long)NB * SS * HH * 48;
        rope_pack_kernel<<<(unsigned)((total + 255) / 256), 256, 0, stream>>>(QKVb, Qb, Kb, VT);
    }
    // 5) attention
    {
        dim3 grid(NB * HH, SS / 64);
        attn_fwd_kernel<<<grid, 128, 0, stream>>>(Qb, Kb, VT, Ob);
    }
    // 6) output projection: (9216,1408) x (1408,1408) + b_o -> fp32 d_out
    {
        dim3 grid(EE / 128, NTOK / 128);
        gemm_bf16_wmma<<<grid, 256, 0, stream>>>(Ob, WoT, b_o, nullptr, out,
                                                 NTOK, EE, EE);
    }
}
